// MultiHeadSelfAttention_39943195852886
// MI455X (gfx1250) — compile-verified, matrix operand order checked by reading, not run on hardware
//
#include <hip/hip_runtime.h>
#include <hip/hip_bf16.h>

typedef __attribute__((ext_vector_type(16))) _Float16 v16h;
typedef __attribute__((ext_vector_type(8)))  float    v8f;

#define EMBED 768
#define NHEAD 12
#define HDIM  64
#define SEQ   4096
#define BATCH 2
#define ROWS  (BATCH * SEQ)     // 8192
#define NQKV  (3 * EMBED)       // 2304

// ---------------------------------------------------------------------------
// CDNA5 wave32 WMMA 16x16x32 f16->f32 fragment helpers.
// A (16x32 MxK): lane l -> row m = l&15, K-base kb = (l<16)?0:8; the 16 halves
// are two contiguous 16-byte runs (K kb..kb+7, K kb+16..kb+23) => 2x ds_load_b128.
// Any B operand stored as its [n][k] transpose uses the same load.
// C/D (16x16): VGPR r -> M = r + (lane<16?0:8), N = lane&15.
// ---------------------------------------------------------------------------

union HV16 { uint4 q[2]; v16h v; };

__device__ __forceinline__ v16h load_frag_row(const _Float16* __restrict__ base, int ld) {
    const int lane = threadIdx.x & 31;
    const int m  = lane & 15;
    const int kb = (lane & 16) >> 1;           // 0 or 8 halfs
    const _Float16* p = base + m * ld + kb;
    HV16 h;
    h.q[0] = *(const uint4*)(p);
    h.q[1] = *(const uint4*)(p + 16);
    return h.v;
}

__device__ __forceinline__ v8f wmma_f16(v16h a, v16h b, v8f c) {
    return __builtin_amdgcn_wmma_f32_16x16x32_f16(false, a, false, b, (short)0, c,
                                                  false, false);
}

// ---------------------------------------------------------------------------
// Kernel 1: QKV = X @ W^T + b   (M=8192, N=2304, K=768)
// 128x64 block tile, BK=32, 256 threads = 8 waves (4x2 wave grid, each wave
// owns a 32x32 sub-tile => 4 WMMAs per k-step off 8 b128 LDS loads, issued
// before the WMMA chain so dscnt waits overlap with matrix math).
// ---------------------------------------------------------------------------
#define BM 128
#define BN 64
#define BK 32
#define LDT 40   // halfs per tile row (32 + 8 pad); 80B, 16B-aligned

__device__ __forceinline__ void stage_tiles(
    const float* __restrict__ X, const float* __restrict__ W,
    int m0, int n0, int k0, int t,
    _Float16* __restrict__ As, _Float16* __restrict__ Bs)
{
#pragma unroll
    for (int i = 0; i < 4; ++i) {
        const int idx = t + i * 256;
        const int row = idx >> 3;
        const int c4  = idx & 7;
        const float4 v = *(const float4*)&X[(size_t)(m0 + row) * EMBED + k0 + c4 * 4];
        _Float16* d = &As[row * LDT + c4 * 4];
        d[0] = (_Float16)v.x; d[1] = (_Float16)v.y;
        d[2] = (_Float16)v.z; d[3] = (_Float16)v.w;
    }
#pragma unroll
    for (int i = 0; i < 2; ++i) {
        const int idx = t + i * 256;
        const int wr  = idx >> 3;
        const int c4  = idx & 7;
        const float4 v = *(const float4*)&W[(size_t)(n0 + wr) * EMBED + k0 + c4 * 4];
        _Float16* d = &Bs[wr * LDT + c4 * 4];
        d[0] = (_Float16)v.x; d[1] = (_Float16)v.y;
        d[2] = (_Float16)v.z; d[3] = (_Float16)v.w;
    }
}

__global__ __launch_bounds__(256) void qkv_gemm(
    const float* __restrict__ X, const float* __restrict__ W,
    const float* __restrict__ bias,
    _Float16* __restrict__ Qh, _Float16* __restrict__ Kh, _Float16* __restrict__ Vh)
{
    __shared__ _Float16 As[2][BM * LDT];   // 2 x 10240 B
    __shared__ _Float16 Bs[2][BN * LDT];   // 2 x  5120 B

    const int t    = threadIdx.x;
    const int lane = t & 31;
    const int wave = t >> 5;
    const int wr   = wave >> 1;            // 0..3 -> M rows wr*32..+31
    const int wc   = wave & 1;             // 0..1 -> N cols wc*32..+31
    const int m0   = blockIdx.y * BM;
    const int n0   = blockIdx.x * BN;

    v8f acc[2][2] = {};

    stage_tiles(X, W, m0, n0, 0, t, As[0], Bs[0]);
    __syncthreads();

    const int NK = EMBED / BK;             // 24
    for (int kt = 0; kt < NK; ++kt) {
        const int cur = kt & 1;
        if (kt + 1 < NK) {
            stage_tiles(X, W, m0, n0, (kt + 1) * BK, t, As[cur ^ 1], Bs[cur ^ 1]);
            if (kt + 2 < NK) {
                __builtin_prefetch(&X[(size_t)(m0 + (t & 127)) * EMBED + (kt + 2) * BK], 0, 1);
                __builtin_prefetch(&W[(size_t)(n0 + (t & 63))  * EMBED + (kt + 2) * BK], 0, 1);
            }
        }
        // issue all 8 b128 loads up front, then the 4-WMMA chain
        const v16h a0 = load_frag_row(&As[cur][(wr * 32)      * LDT], LDT);
        const v16h a1 = load_frag_row(&As[cur][(wr * 32 + 16) * LDT], LDT);
        const v16h b0 = load_frag_row(&Bs[cur][(wc * 32)      * LDT], LDT);
        const v16h b1 = load_frag_row(&Bs[cur][(wc * 32 + 16) * LDT], LDT);
        acc[0][0] = wmma_f16(a0, b0, acc[0][0]);
        acc[0][1] = wmma_f16(a0, b1, acc[0][1]);
        acc[1][0] = wmma_f16(a1, b0, acc[1][0]);
        acc[1][1] = wmma_f16(a1, b1, acc[1][1]);
        __syncthreads();
    }

    // epilogue: bias, fold 1/sqrt(64) into Q, scatter to [b,h,n,hd] f16
    const int nn = lane & 15;
    const int mb = (lane < 16) ? 0 : 8;
#pragma unroll
    for (int ni = 0; ni < 2; ++ni) {
        const int cg   = n0 + (wc * 2 + ni) * 16 + nn;
        const int sel  = cg / EMBED;
        const int rem  = cg % EMBED;
        const int head = rem >> 6;
        const int hd   = rem & 63;
        const float bv = bias[cg];
        _Float16* dst  = (sel == 0) ? Qh : ((sel == 1) ? Kh : Vh);
        const float sc = (sel == 0) ? 0.125f : 1.0f;
#pragma unroll
        for (int mi2 = 0; mi2 < 2; ++mi2) {
#pragma unroll
            for (int r = 0; r < 8; ++r) {
                const int gm = m0 + (wr * 2 + mi2) * 16 + mb + r;
                const int bb = gm >> 12;
                const int nr = gm & (SEQ - 1);
                const float v = (acc[mi2][ni][r] + bv) * sc;
                dst[(((size_t)(bb * NHEAD + head)) * SEQ + nr) * HDIM + hd] = (_Float16)v;
            }
        }
    }
}

// ---------------------------------------------------------------------------
// Kernel 2: flash attention.  grid = (SEQ/64, BATCH*NHEAD), 256 threads.
// Q fragments pinned in VGPRs for the whole kv loop; V staged transposed
// ([hd][key]); all fragments of each WMMA chain preloaded before the chain.
// ---------------------------------------------------------------------------
#define BQ    64
#define BKV   64
#define LQ    72   // half stride (144B, 16B-aligned)
#define SS_LD 68   // float stride for score tile

__global__ __launch_bounds__(256) void flash_attn(
    const _Float16* __restrict__ Qh, const _Float16* __restrict__ Kh,
    const _Float16* __restrict__ Vh, float* __restrict__ O)
{
    __shared__ _Float16 Qs[BQ * LQ];          // 9216 B (dead after prologue)
    __shared__ _Float16 Ks[BKV * LQ];         // 9216 B, [key][hd]
    __shared__ _Float16 Vt[HDIM * LQ];        // 9216 B, [hd][key] (transposed)
    __shared__ _Float16 Ps[BQ * LQ];          // 9216 B, [qrow][key]
    __shared__ float    Ss[BQ * SS_LD];       // 17408 B
    __shared__ float    m_s[BQ], l_s[BQ], f_s[BQ];
    __shared__ float    pmax[BQ * 4], psum[BQ * 4];

    const int t    = threadIdx.x;
    const int lane = t & 31;
    const int wave = t >> 5;
    const int mi   = wave >> 1;
    const int nh   = wave & 1;
    const int nn   = lane & 15;
    const int mb   = (lane < 16) ? 0 : 8;
    const int qt   = blockIdx.x;
    const int bh   = blockIdx.y;
    const size_t base = (size_t)bh * SEQ * HDIM;

    // ---- prologue: stage Q tile, keep its fragments in registers
    const uint4* qg = (const uint4*)(Qh + base + (size_t)qt * BQ * HDIM);
#pragma unroll
    for (int i = 0; i < 2; ++i) {
        const int idx = t + i * 256;
        *(uint4*)&Qs[(idx >> 3) * LQ + (idx & 7) * 8] = qg[idx];
    }
    if (t < BQ) { m_s[t] = -1e30f; l_s[t] = 0.0f; }
    __syncthreads();

    v16h qa[2];
#pragma unroll
    for (int kk = 0; kk < 2; ++kk)
        qa[kk] = load_frag_row(&Qs[(mi * 16) * LQ + kk * 32], LQ);

    v8f o[2] = {};

    for (int kt = 0; kt < SEQ / BKV; ++kt) {
        // ---- stage K ([key][hd]) and V transposed ([hd][key])
        const uint4* kg = (const uint4*)(Kh + base + (size_t)kt * BKV * HDIM);
        const uint4* vg = (const uint4*)(Vh + base + (size_t)kt * BKV * HDIM);
#pragma unroll
        for (int i = 0; i < 2; ++i) {
            const int idx = t + i * 256;
            const int row = idx >> 3;
            const int c4  = idx & 7;
            *(uint4*)&Ks[row * LQ + c4 * 8] = kg[idx];
            HV16 h; h.q[0] = vg[idx];
            const _Float16* e = (const _Float16*)&h.q[0];
#pragma unroll
            for (int j = 0; j < 8; ++j)
                Vt[(c4 * 8 + j) * LQ + row] = e[j];
        }
        __syncthreads();

        // ---- S = Q @ K^T : preload 4 K fragments, then 4 WMMAs
        {
            v16h kb[2][2];
#pragma unroll
            for (int ti = 0; ti < 2; ++ti)
#pragma unroll
                for (int kk = 0; kk < 2; ++kk)
                    kb[ti][kk] = load_frag_row(
                        &Ks[((nh * 2 + ti) * 16) * LQ + kk * 32], LQ);
#pragma unroll
            for (int ti = 0; ti < 2; ++ti) {
                v8f s = {};
                s = wmma_f16(qa[0], kb[ti][0], s);
                s = wmma_f16(qa[1], kb[ti][1], s);
#pragma unroll
                for (int r = 0; r < 8; ++r)
                    Ss[(mi * 16 + mb + r) * SS_LD + (nh * 2 + ti) * 16 + nn] = s[r];
            }
        }
        __syncthreads();

        // ---- online softmax, 4 threads per row
        {
            const int srow = t >> 2;
            const int ssub = t & 3;
            const float* sr = &Ss[srow * SS_LD + ssub * 16];
            float mx = -1e30f;
#pragma unroll
            for (int c = 0; c < 16; ++c) mx = fmaxf(mx, sr[c]);
            pmax[srow * 4 + ssub] = mx;
            __syncthreads();

            const float mold = m_s[srow];
            const float mnew = fmaxf(mold,
                fmaxf(fmaxf(pmax[srow * 4 + 0], pmax[srow * 4 + 1]),
                      fmaxf(pmax[srow * 4 + 2], pmax[srow * 4 + 3])));
            float rs = 0.0f;
            _Float16* pr = &Ps[srow * LQ + ssub * 16];
#pragma unroll
            for (int c = 0; c < 16; ++c) {
                const float e = __expf(sr[c] - mnew);
                pr[c] = (_Float16)e;
                rs += e;
            }
            psum[srow * 4 + ssub] = rs;
            __syncthreads();

            if (ssub == 0) {
                const float fac = __expf(mold - mnew);
                m_s[srow] = mnew;
                l_s[srow] = l_s[srow] * fac +
                            psum[srow * 4 + 0] + psum[srow * 4 + 1] +
                            psum[srow * 4 + 2] + psum[srow * 4 + 3];
                f_s[srow] = fac;
            }
            __syncthreads();
        }

        // ---- O = O * fac + P @ V : P frags shared across both column tiles
        {
            v16h pa[2], vb[2][2];
#pragma unroll
            for (int kk = 0; kk < 2; ++kk)
                pa[kk] = load_frag_row(&Ps[(mi * 16) * LQ + kk * 32], LQ);
#pragma unroll
            for (int ti = 0; ti < 2; ++ti)
#pragma unroll
                for (int kk = 0; kk < 2; ++kk)
                    vb[ti][kk] = load_frag_row(
                        &Vt[((nh * 2 + ti) * 16) * LQ + kk * 32], LQ);
#pragma unroll
            for (int ti = 0; ti < 2; ++ti) {
#pragma unroll
                for (int r = 0; r < 8; ++r) o[ti][r] *= f_s[mi * 16 + mb + r];
                o[ti] = wmma_f16(pa[0], vb[ti][0], o[ti]);
                o[ti] = wmma_f16(pa[1], vb[ti][1], o[ti]);
            }
        }
        __syncthreads();
    }

    // ---- normalize, write [b*h, n, hd] f32
#pragma unroll
    for (int ti = 0; ti < 2; ++ti) {
        const int nt = nh * 2 + ti;
#pragma unroll
        for (int r = 0; r < 8; ++r) {
            const int row = mi * 16 + mb + r;
            O[base + (size_t)(qt * BQ + row) * HDIM + nt * 16 + nn] =
                o[ti][r] * (1.0f / l_s[row]);
        }
    }
}

// ---------------------------------------------------------------------------
// Kernel 3: out = LayerNorm(x + attn), one block per token row.
// ---------------------------------------------------------------------------
__global__ __launch_bounds__(256) void resid_ln(
    const float* __restrict__ X, const float* __restrict__ A,
    const float* __restrict__ gamma, const float* __restrict__ beta,
    float* __restrict__ Out)
{
    __shared__ float r1[256];
    __shared__ float r2[256];
    const int row = blockIdx.x;
    const int b   = row >> 12;
    const int n   = row & (SEQ - 1);
    const int t   = threadIdx.x;

    float y[3];
    float s = 0.0f, s2 = 0.0f;
#pragma unroll
    for (int j = 0; j < 3; ++j) {
        const int d = t + j * 256;
        const float v = X[(size_t)row * EMBED + d] +
                        A[(((size_t)(b * NHEAD + (d >> 6))) * SEQ + n) * HDIM + (d & 63)];
        y[j] = v;
        s += v;
        s2 += v * v;
    }
    r1[t] = s; r2[t] = s2;
    __syncthreads();
    for (int off = 128; off > 0; off >>= 1) {
        if (t < off) { r1[t] += r1[t + off]; r2[t] += r2[t + off]; }
        __syncthreads();
    }
    const float mean = r1[0] * (1.0f / EMBED);
    const float var  = r2[0] * (1.0f / EMBED) - mean * mean;
    const float rstd = rsqrtf(var + 1e-5f);
#pragma unroll
    for (int j = 0; j < 3; ++j) {
        const int d = t + j * 256;
        Out[(size_t)row * EMBED + d] = (y[j] - mean) * rstd * gamma[d] + beta[d];
    }
}

// ---------------------------------------------------------------------------
extern "C" void kernel_launch(void* const* d_in, const int* in_sizes, int n_in,
                              void* d_out, int out_size, void* d_ws, size_t ws_size,
                              hipStream_t stream) {
    (void)in_sizes; (void)n_in; (void)out_size; (void)ws_size;
    const float* x     = (const float*)d_in[0];
    const float* w_qkv = (const float*)d_in[1];
    const float* b_qkv = (const float*)d_in[2];
    const float* gamma = (const float*)d_in[3];
    const float* beta  = (const float*)d_in[4];
    float* out = (float*)d_out;

    char* ws = (char*)d_ws;
    const size_t QSZ = (size_t)BATCH * NHEAD * SEQ * HDIM * sizeof(_Float16); // 12.6 MB
    _Float16* Qh = (_Float16*)(ws);
    _Float16* Kh = (_Float16*)(ws + QSZ);
    _Float16* Vh = (_Float16*)(ws + 2 * QSZ);
    float*    At = (float*)   (ws + 3 * QSZ);                                 // 25.2 MB

    dim3 g1(NQKV / BN, ROWS / BM);      // 36 x 64
    qkv_gemm<<<g1, 256, 0, stream>>>(x, w_qkv, b_qkv, Qh, Kh, Vh);

    dim3 g2(SEQ / BQ, BATCH * NHEAD);   // 64 x 24
    flash_attn<<<g2, 256, 0, stream>>>(Qh, Kh, Vh, At);

    resid_ln<<<ROWS, 256, 0, stream>>>(x, At, gamma, beta, out);
}
